// ProbAttention_23261542875384
// MI455X (gfx1250) — matrix operand order repaired and checked by reading, not compiled, not run on hardware
//
#include <hip/hip_runtime.h>
#include <hip/hip_bf16.h>

// ---------------------------------------------------------------------------
// ProbSparse attention (Informer) for MI455X / gfx1250.
// B=2, H=8, L=4096, D=64, sample_k = top_k = 40 (5*int(log(4096))).
// Memory-bound problem (~2 GFLOP vs ~67 MB => 2.9us HBM floor), so the GEMM
// phases use f16 WMMA (v_wmma_f32_16x16x32_f16) with f32 accumulation.
// ---------------------------------------------------------------------------

#define BH   16        // B*H
#define LQ   4096
#define DD   64
#define SK   40        // sample_k
#define TK   40        // top_k
#define TP   48        // top_k padded to 3 WMMA tiles
#define LD   (LQ*DD)   // 262144 elements per (b,h)

typedef __attribute__((ext_vector_type(16))) _Float16 v16h;
typedef __attribute__((ext_vector_type(8)))  _Float16 v8h;
typedef __attribute__((ext_vector_type(8)))  float    v8f;

union U16h { v16h v; v8h h[2]; _Float16 e[16]; };
union U8f  { v8f  v; float e[8]; };

// ---------------------------------------------------------------------------
// 1) M[bh][l] = max_s(q . k_s) - mean_s(q . k_s) over 40 sampled keys.
//    One wave32 per query; each lane owns 2 of the 64 dims; butterfly reduce.
// ---------------------------------------------------------------------------
__global__ void m_kernel(const float* __restrict__ Q, const float* __restrict__ K,
                         const int* __restrict__ samp, float* __restrict__ M) {
    const int wid  = threadIdx.x >> 5;
    const int lane = threadIdx.x & 31;
    const int gq   = blockIdx.x * 8 + wid;          // 0 .. BH*LQ-1
    const int bh   = gq >> 12;                      // / 4096
    const int l    = gq & 4095;
    const float2 qv = *(const float2*)(Q + (size_t)bh * LD + (size_t)l * DD + 2 * lane);
    const float* kb = K + (size_t)bh * LD;
    float mx = -INFINITY, sm = 0.0f;
    int idx = samp[l * SK];
    for (int s = 0; s < SK; ++s) {
        const float2 kv = *(const float2*)(kb + (size_t)idx * DD + 2 * lane);
        if (s + 1 < SK) {
            idx = samp[l * SK + s + 1];
            __builtin_prefetch(kb + (size_t)idx * DD + 2 * lane, 0, 3);
        }
        float p = qv.x * kv.x + qv.y * kv.y;
#pragma unroll
        for (int off = 16; off >= 1; off >>= 1) p += __shfl_xor(p, off, 32);
        mx = fmaxf(mx, p);
        sm += p;
    }
    if (lane == 0) M[(size_t)bh * LQ + l] = mx - sm * (1.0f / SK);
}

// ---------------------------------------------------------------------------
// 2) Iterative top-40 argmax per (b,h). Mutates M (rewritten by m_kernel each
//    call, so graph replays stay deterministic).
// ---------------------------------------------------------------------------
__global__ void topk_kernel(float* __restrict__ M, int* __restrict__ topidx) {
    const int bh  = blockIdx.x;
    const int tid = threadIdx.x;
    float* Mrow = M + (size_t)bh * LQ;
    __shared__ float sv[256];
    __shared__ int   si[256];
    for (int t = 0; t < TK; ++t) {
        float bv = -INFINITY; int bi = 0;
        for (int l = tid; l < LQ; l += 256) {
            float v = Mrow[l];
            if (v > bv) { bv = v; bi = l; }
        }
        sv[tid] = bv; si[tid] = bi;
        __syncthreads();
        for (int s = 128; s > 0; s >>= 1) {
            if (tid < s) {
                if (sv[tid + s] > sv[tid] ||
                    (sv[tid + s] == sv[tid] && si[tid + s] < si[tid])) {
                    sv[tid] = sv[tid + s]; si[tid] = si[tid + s];
                }
            }
            __syncthreads();
        }
        if (tid == 0) { topidx[bh * TK + t] = si[0]; Mrow[si[0]] = -INFINITY; }
        __syncthreads();
    }
}

// ---------------------------------------------------------------------------
// 3) Kh = f16(K) (row-major, B-fragment-ready for QK^T), and
//    Vth = f16(V) transposed to (bh, D, L) so PV B-fragments are contiguous.
// ---------------------------------------------------------------------------
__global__ void convert_kernel(const float* __restrict__ K, const float* __restrict__ V,
                               _Float16* __restrict__ Kh, _Float16* __restrict__ Vth) {
    const size_t stride = (size_t)gridDim.x * blockDim.x;
    const size_t N = (size_t)BH * LD;
    for (size_t i = blockIdx.x * (size_t)blockDim.x + threadIdx.x; i < N; i += stride)
        Kh[i] = (_Float16)K[i];
    for (size_t j = blockIdx.x * (size_t)blockDim.x + threadIdx.x; j < N; j += stride) {
        const size_t bh = j >> 18;          // / (D*L)
        const size_t r  = j & (LD - 1);
        const size_t d  = r >> 12;          // / L
        const size_t l  = r & (LQ - 1);
        Vth[j] = (_Float16)V[(bh << 18) + l * DD + d];
    }
}

// ---------------------------------------------------------------------------
// 4) Vmean[bh][d] = mean over L of V.
// ---------------------------------------------------------------------------
__global__ void vmean_kernel(const float* __restrict__ V, float* __restrict__ vmean) {
    const int bh = blockIdx.x, tid = threadIdx.x;
    const int d = tid & 63, part = tid >> 6;
    __shared__ float red[256];
    float s = 0.0f;
    for (int l = part; l < LQ; l += 4) s += V[(size_t)bh * LD + (size_t)l * DD + d];
    red[tid] = s;
    __syncthreads();
    if (part == 0)
        vmean[bh * DD + d] =
            (red[d] + red[64 + d] + red[128 + d] + red[192 + d]) * (1.0f / LQ);
}

// ---------------------------------------------------------------------------
// 5) Gather top-40 Q rows -> f16, zero-padded to 48 rows.
// ---------------------------------------------------------------------------
__global__ void qred_kernel(const float* __restrict__ Q, const int* __restrict__ topidx,
                            _Float16* __restrict__ Qredh) {
    const int bh = blockIdx.x;
    for (int e = threadIdx.x; e < TP * DD; e += 256) {
        const int row = e >> 6, d = e & 63;
        _Float16 val = (_Float16)0.0f;
        if (row < TK) {
            const int idx = topidx[bh * TK + row];
            val = (_Float16)Q[(size_t)bh * LD + (size_t)idx * DD + d];
        }
        Qredh[bh * TP * DD + e] = val;
    }
}

// ---------------------------------------------------------------------------
// 6) scores = (Q_red . K^T) * 0.125 via v_wmma_f32_16x16x32_f16, stored f16.
//    Grid: 16 bh * 16 n-spans; 8 waves/block, each wave owns (m,n) tiles.
//    A fragment (16x32 f16): lane = (g=lane>>4, r=lane&15);
//      e[0..7]=K kb+16g.., e[8..15]=K kb+16g+8..  -> two aligned 16B loads.
//    B fragment (32x16 f16): col = lane&15, K = 16g + e  -> one 32B load
//      straight from a K row (dims are contiguous in memory).
// ---------------------------------------------------------------------------
__global__ void scores_kernel(const _Float16* __restrict__ Qredh,
                              const _Float16* __restrict__ Kh,
                              _Float16* __restrict__ Sh) {
    const int bh   = blockIdx.x >> 4;
    const int nb   = blockIdx.x & 15;
    const int wid  = threadIdx.x >> 5;
    const int lane = threadIdx.x & 31;
    const int lm   = lane & 15;
    const int g    = lane >> 4;
    const _Float16* qb = Qredh + bh * TP * DD;
    const _Float16* kb = Kh + (size_t)bh * LD;
    _Float16* sb = Sh + (size_t)bh * TP * LQ;
    for (int tt = wid; tt < 48; tt += 8) {
        const int mtile = tt % 3;
        const int ntile = nb * 16 + tt / 3;
        const int r = mtile * 16 + lm;
        const int n = ntile * 16 + lm;
        U8f c = {};
#pragma unroll
        for (int ks = 0; ks < 2; ++ks) {
            const int kbs = ks * 32;
            U16h a, b;
            a.h[0] = *(const v8h*)(qb + r * DD + kbs + g * 16);
            a.h[1] = *(const v8h*)(qb + r * DD + kbs + g * 16 + 8);
            b.v    = *(const v16h*)(kb + (size_t)n * DD + kbs + g * 16);
            c.v = __builtin_amdgcn_wmma_f32_16x16x32_f16(
                false, a.v, false, b.v, (short)0, c.v, false, false);
        }
        const int col = ntile * 16 + lm;
#pragma unroll
        for (int i = 0; i < 8; ++i) {
            const int row = mtile * 16 + g * 8 + i;
            sb[(size_t)row * LQ + col] = (_Float16)(c.e[i] * 0.125f);
        }
    }
}

// ---------------------------------------------------------------------------
// 7) Row softmax in place on Sh; pad rows (>=40) are zeroed so the padded
//    WMMA M-tiles of the PV matmul contribute nothing.
// ---------------------------------------------------------------------------
__global__ void softmax_kernel(_Float16* __restrict__ Sh) {
    const int bh = blockIdx.x / TP, r = blockIdx.x % TP;
    const int tid = threadIdx.x;
    _Float16* row = Sh + ((size_t)bh * TP + r) * LQ;
    if (r >= TK) {
#pragma unroll
        for (int j = 0; j < 16; ++j) row[tid + 256 * j] = (_Float16)0.0f;
        return;
    }
    float v[16];
    float mx = -INFINITY;
#pragma unroll
    for (int j = 0; j < 16; ++j) {
        v[j] = (float)row[tid + 256 * j];
        mx = fmaxf(mx, v[j]);
    }
    __shared__ float red[256];
    red[tid] = mx;
    __syncthreads();
    for (int s = 128; s > 0; s >>= 1) {
        if (tid < s) red[tid] = fmaxf(red[tid], red[tid + s]);
        __syncthreads();
    }
    mx = red[0];
    __syncthreads();
    float sm = 0.0f;
#pragma unroll
    for (int j = 0; j < 16; ++j) { v[j] = __expf(v[j] - mx); sm += v[j]; }
    red[tid] = sm;
    __syncthreads();
    for (int s = 128; s > 0; s >>= 1) {
        if (tid < s) red[tid] += red[tid + s];
        __syncthreads();
    }
    const float inv = 1.0f / red[0];
#pragma unroll
    for (int j = 0; j < 16; ++j) row[tid + 256 * j] = (_Float16)(v[j] * inv);
}

// ---------------------------------------------------------------------------
// 8) ctx = attn . V via WMMA. A from Sh (rows contiguous), B from transposed
//    f16 V (each lane's 16 K-values contiguous -> one aligned 32B load).
// ---------------------------------------------------------------------------
__global__ void pv_kernel(const _Float16* __restrict__ Sh,
                          const _Float16* __restrict__ Vth,
                          float* __restrict__ ctx) {
    const int bh   = blockIdx.x;
    const int wid  = threadIdx.x >> 5;
    const int lane = threadIdx.x & 31;
    const int lm   = lane & 15;
    const int g    = lane >> 4;
    const _Float16* ab = Sh + (size_t)bh * TP * LQ;
    const _Float16* vb = Vth + (size_t)bh * LD;
    for (int tt = wid; tt < 12; tt += 8) {
        const int mtile = tt % 3;
        const int ntile = tt / 3;
        const int r = mtile * 16 + lm;   // attn row
        const int n = ntile * 16 + lm;   // output dim
        U8f c = {};
        for (int ks = 0; ks < 128; ++ks) {
            const int kbs = ks * 32;
            U16h a, b;
            a.h[0] = *(const v8h*)(ab + (size_t)r * LQ + kbs + g * 16);
            a.h[1] = *(const v8h*)(ab + (size_t)r * LQ + kbs + g * 16 + 8);
            b.v    = *(const v16h*)(vb + (size_t)n * LQ + kbs + g * 16);
            c.v = __builtin_amdgcn_wmma_f32_16x16x32_f16(
                false, a.v, false, b.v, (short)0, c.v, false, false);
        }
#pragma unroll
        for (int i = 0; i < 8; ++i) {
            const int row = mtile * 16 + g * 8 + i;
            const int col = ntile * 16 + lm;
            ctx[(size_t)bh * TP * DD + (size_t)row * DD + col] = c.e[i];
        }
    }
}

// ---------------------------------------------------------------------------
// 9) out = broadcast Vmean (float4 stores), then 10) scatter context rows.
// ---------------------------------------------------------------------------
__global__ void bcast_kernel(const float* __restrict__ vmean, float* __restrict__ out) {
    const int i  = blockIdx.x * 256 + threadIdx.x;   // over BH*L*D/4
    const int bh = i >> 16;                           // / (L*D/4)
    const int c4 = i & 15;
    ((float4*)out)[i] = ((const float4*)vmean)[bh * 16 + c4];
}

__global__ void scatter_kernel(const float* __restrict__ ctx,
                               const int* __restrict__ topidx,
                               float* __restrict__ out) {
    const int bh = blockIdx.x;
    for (int e = threadIdx.x; e < TK * DD; e += 256) {
        const int row = e >> 6, d = e & 63;
        const int idx = topidx[bh * TK + row];
        out[(size_t)bh * LD + (size_t)idx * DD + d] =
            ctx[(size_t)bh * TP * DD + (size_t)row * DD + d];
    }
}

// ---------------------------------------------------------------------------
// Launch
// ---------------------------------------------------------------------------
extern "C" void kernel_launch(void* const* d_in, const int* in_sizes, int n_in,
                              void* d_out, int out_size, void* d_ws, size_t ws_size,
                              hipStream_t stream) {
    const float* Q = (const float*)d_in[0];
    const float* K = (const float*)d_in[1];
    const float* V = (const float*)d_in[2];
    const int* samp = (const int*)d_in[3];
    float* out = (float*)d_out;

    char* ws = (char*)d_ws;
    float*    Mbuf   = (float*)(ws + 0);            //   262144 B
    int*      topidx = (int*)(ws + 262144);          //     4096 B (padded)
    _Float16* Kh     = (_Float16*)(ws + 266240);     //  8388608 B
    _Float16* Vth    = (_Float16*)(ws + 8654848);    //  8388608 B
    _Float16* Qredh  = (_Float16*)(ws + 17043456);   //    98304 B
    _Float16* Sh     = (_Float16*)(ws + 17141760);   //  6291456 B
    float*    ctx    = (float*)(ws + 23433216);      //   196608 B
    float*    vmean  = (float*)(ws + 23629824);      //     4096 B  (total ~22.5 MB)

    m_kernel<<<BH * LQ / 8, 256, 0, stream>>>(Q, K, samp, Mbuf);
    topk_kernel<<<BH, 256, 0, stream>>>(Mbuf, topidx);
    convert_kernel<<<2048, 256, 0, stream>>>(K, V, Kh, Vth);
    vmean_kernel<<<BH, 256, 0, stream>>>(V, vmean);
    qred_kernel<<<BH, 256, 0, stream>>>(Q, topidx, Qredh);
    scores_kernel<<<BH * 16, 256, 0, stream>>>(Qredh, Kh, Sh);
    softmax_kernel<<<BH * TP, 256, 0, stream>>>(Sh);
    pv_kernel<<<BH, 256, 0, stream>>>(Sh, Vth, ctx);
    bcast_kernel<<<BH * LQ * DD / 4 / 256, 256, 0, stream>>>(vmean, out);
    scatter_kernel<<<BH, 256, 0, stream>>>(ctx, topidx, out);
}
